// GCN_32727650795882
// MI455X (gfx1250) — matrix operand order repaired and checked by reading, not compiled, not run on hardware
//
#include <hip/hip_runtime.h>
#include <stdint.h>

// ---------------------------------------------------------------------------
// CDNA5 WMMA types (wave32)
// ---------------------------------------------------------------------------
typedef __bf16 v16bf __attribute__((ext_vector_type(16)));
typedef float  v8f   __attribute__((ext_vector_type(8)));
typedef unsigned int v8u __attribute__((ext_vector_type(8)));
typedef int v4i __attribute__((ext_vector_type(4)));

#define GLOBAL_AS __attribute__((address_space(1)))
#define LDS_AS    __attribute__((address_space(3)))

// ---- async global->LDS path (gfx1250); fall back to sync copy if absent ----
#if __has_builtin(__builtin_amdgcn_global_load_async_to_lds_b128)
#define HAVE_ASYNC_LDS 1
#else
#define HAVE_ASYNC_LDS 0
#endif

#if HAVE_ASYNC_LDS
#if __has_builtin(__builtin_amdgcn_s_wait_asynccnt)
#define WAIT_ASYNC(n) __builtin_amdgcn_s_wait_asynccnt(n)
#else
#define WAIT_ASYNC(n) asm volatile("s_wait_asynccnt " #n ::: "memory")
#endif
#else
#define WAIT_ASYNC(n)
#endif

__device__ __forceinline__ void copy16_to_lds(const unsigned short* g,
                                              unsigned short* l) {
#if HAVE_ASYNC_LDS
  // signature (from hipcc diagnostic): (v4i AS1* src, v4i AS3* dst, Ii, Ii)
  __builtin_amdgcn_global_load_async_to_lds_b128(
      (GLOBAL_AS v4i*)g, (LDS_AS v4i*)l, /*offset=*/0, /*cpol=*/0);
#else
  *(uint4*)l = *(const uint4*)g;   // sync fallback: global_load_b128 + ds_store
#endif
}

__device__ __forceinline__ unsigned short f2bf(float f) {
  unsigned u = __builtin_bit_cast(unsigned, f);
  u += 0x7FFFu + ((u >> 16) & 1u);        // round-to-nearest-even
  return (unsigned short)(u >> 16);
}

#define BM 128
#define BN 64
#define BK 32

// ---------------------------------------------------------------------------
// Double-buffered bf16 WMMA GEMM: Out[M x Ncols] = A[M x K] * B[K x Ncols]
// A, B are bf16 row-major. 256 threads = 8 waves; wave (wr,wc) owns a 32x32
// output patch -> 4 v_wmma_f32_16x16x32_bf16 per K-step. Tiles are fetched
// with GLOBAL_LOAD_ASYNC_TO_LDS_B128 one iteration ahead (3 per-wave async
// ops per tile -> s_wait_asynccnt 3 keeps the pipeline one tile deep).
//   FUSED: epilogue relu(acc + x[m]*Wa[n] + c[m]*Wc[n] + bl[n]) -> bf16 out
//   else : acc -> fp32 out
// ---------------------------------------------------------------------------
template <bool FUSED>
__global__ __launch_bounds__(256)
void wmma_gemm_bf16(const unsigned short* __restrict__ A,
                    const unsigned short* __restrict__ B,
                    int M, int K, int Ncols,
                    const float* __restrict__ xv, const float* __restrict__ cv,
                    const float* __restrict__ Wa, const float* __restrict__ Wc,
                    const float* __restrict__ bias,
                    void* __restrict__ Outv)
{
  __shared__ unsigned short ldsA[2][BM * BK];   // 2 x 8 KB
  __shared__ unsigned short ldsB[2][BK * BN];   // 2 x 4 KB

  const int tid  = threadIdx.x;
  const int lane = tid & 31;
  const int wave = tid >> 5;
  const int wr   = wave >> 1;   // 0..3 : 32-row group
  const int wc   = wave & 1;    // 0..1 : 32-col group
  const int half = lane >> 4;
  const int m0   = blockIdx.y * BM;
  const int n0   = blockIdx.x * BN;

  // Per-thread tile-fetch coordinates (3 x 16B per thread per tile).
  // A tile: 512 chunks of 8 bf16; B tile: 256 chunks of 8 bf16.
  const int ca0_row = (tid)        >> 2, ca0_seg = (tid)        & 3;
  const int ca1_row = (tid + 256)  >> 2, ca1_seg = (tid + 256)  & 3;
  const int cb_row  = tid >> 3,          cb_seg  = tid & 7;
  const int gmA0 = (m0 + ca0_row < M) ? (m0 + ca0_row) : (M - 1);
  const int gmA1 = (m0 + ca1_row < M) ? (m0 + ca1_row) : (M - 1);

  v8f acc[4];
  #pragma unroll
  for (int t = 0; t < 4; ++t) {
    #pragma unroll
    for (int i = 0; i < 8; ++i) acc[t][i] = 0.0f;
  }

  const int nk = K / BK;

  // issue tile `ki` into buffer `buf`
  auto issue = [&](int ki, int buf) {
    const int kc = ki * BK;
    copy16_to_lds(A + (size_t)gmA0 * K + kc + ca0_seg * 8,
                  &ldsA[buf][ca0_row * BK + ca0_seg * 8]);
    copy16_to_lds(A + (size_t)gmA1 * K + kc + ca1_seg * 8,
                  &ldsA[buf][ca1_row * BK + ca1_seg * 8]);
    copy16_to_lds(B + (size_t)(kc + cb_row) * Ncols + n0 + cb_seg * 8,
                  &ldsB[buf][cb_row * BN + cb_seg * 8]);
  };

  issue(0, 0);
  for (int i = 0; i < nk; ++i) {
    const int cur = i & 1;
    if (i + 1 < nk) {
      issue(i + 1, cur ^ 1);   // buffer cur^1 was released by last iteration's
      WAIT_ASYNC(3);           // trailing barrier; wait for tile i only
    } else {
      WAIT_ASYNC(0);
    }
    __syncthreads();

    // ---- fragments per CDNA5 layouts, 4 WMMAs ----
    v16bf afr[2], bfr[2];
    #pragma unroll
    for (int t = 0; t < 2; ++t) {
      // A 16x32: lane&15 -> M; element e -> K = (e>>3)*16 + half*8 + (e&7)
      const int arow = 32 * wr + 16 * t + (lane & 15);
      v8u aw;
      #pragma unroll
      for (int j = 0; j < 8; ++j) {
        const int kk = ((j >> 2) << 4) + (half << 3) + ((j & 3) << 1);
        aw[j] = *(const unsigned int*)&ldsA[cur][arow * BK + kk];
      }
      afr[t] = __builtin_bit_cast(v16bf, aw);
      // B 32x16: lane&15 -> N; element e -> K = (e>>3)*16 + half*8 + (e&7)
      const int bcol = 32 * wc + 16 * t + (lane & 15);
      v8u bw;
      #pragma unroll
      for (int j = 0; j < 8; ++j) {
        const int kk = ((j >> 2) << 4) + (half << 3) + ((j & 3) << 1);
        bw[j] = (unsigned int)ldsB[cur][kk * BN + bcol] |
                ((unsigned int)ldsB[cur][(kk + 1) * BN + bcol] << 16);
      }
      bfr[t] = __builtin_bit_cast(v16bf, bw);
    }
    #pragma unroll
    for (int ta = 0; ta < 2; ++ta)
      #pragma unroll
      for (int tb = 0; tb < 2; ++tb)
        acc[ta * 2 + tb] = __builtin_amdgcn_wmma_f32_16x16x32_bf16(
            false, afr[ta], false, bfr[tb], (short)0, acc[ta * 2 + tb],
            false, false);
    __syncthreads();   // release buffer `cur` for the tile after next
  }

  // ---- epilogue: C/D layout VGPR v, lane l -> M = v + 8*(l>>4), N = l&15 ----
  #pragma unroll
  for (int ta = 0; ta < 2; ++ta) {
    #pragma unroll
    for (int tb = 0; tb < 2; ++tb) {
      const int n = n0 + 32 * wc + 16 * tb + (lane & 15);
      float wan = 0.f, wcn = 0.f, bn = 0.f;
      if constexpr (FUSED) { wan = Wa[n]; wcn = Wc[n]; bn = bias[n]; }
      const v8f a = acc[ta * 2 + tb];
      #pragma unroll
      for (int v = 0; v < 8; ++v) {
        const int m = m0 + 32 * wr + 16 * ta + v + 8 * half;
        if (m < M) {
          float val = a[v];
          if constexpr (FUSED) {
            val += xv[m] * wan + cv[m] * wcn + bn;
            val = val > 0.f ? val : 0.f;
            ((unsigned short*)Outv)[(size_t)m * Ncols + n] = f2bf(val);
          } else {
            ((float*)Outv)[(size_t)m * Ncols + n] = val;
          }
        }
      }
    }
  }
}

// ---------------------------------------------------------------------------
// Elementwise / graph helpers
// ---------------------------------------------------------------------------
__global__ void f32_to_bf16_kernel(const float* __restrict__ in,
                                   unsigned short* __restrict__ out,
                                   long long n4) {
  long long i = (long long)blockIdx.x * blockDim.x + threadIdx.x;
  if (i < n4) {
    float4 f = ((const float4*)in)[i];
    ushort4 o;
    o.x = f2bf(f.x); o.y = f2bf(f.y); o.z = f2bf(f.z); o.w = f2bf(f.w);
    ((ushort4*)out)[i] = o;
  }
}

__global__ void set_const_kernel(float* p, float v, int n) {
  int i = blockIdx.x * blockDim.x + threadIdx.x;
  if (i < n) p[i] = v;
}

__global__ void deg_edges_kernel(const int* __restrict__ dst,
                                 const float* __restrict__ w,
                                 float* deg, int E) {
  int e = blockIdx.x * blockDim.x + threadIdx.x;
  if (e < E) atomicAdd(&deg[dst[e]], w[e]);
}

__global__ void deg_fin_kernel(float* d, int n) {
  int i = blockIdx.x * blockDim.x + threadIdx.x;
  if (i < n) {
    float v = d[i];
    d[i] = v > 0.f ? rsqrtf(v) : 0.f;
  }
}

// out[i,:] = dinv[i]^2 * t[i,:]   (self-loop term; initializes out)
__global__ void self_init_kernel(const float* __restrict__ t,
                                 const float* __restrict__ dinv,
                                 float* __restrict__ o, int n, int F) {
  int idx = blockIdx.x * blockDim.x + threadIdx.x;
  if (idx < n * F) {
    int i = idx / F;
    float s = dinv[i];
    o[idx] = s * s * t[idx];
  }
}

// one wave32 per edge; lane-parallel over features; fp32 global atomics
__global__ void edge_agg_kernel(const int* __restrict__ src,
                                const int* __restrict__ dst,
                                const float* __restrict__ w,
                                const float* __restrict__ dinv,
                                const float* __restrict__ t,
                                float* __restrict__ o, int E, int F) {
  int g = blockIdx.x * blockDim.x + threadIdx.x;
  int e = g >> 5;
  int lane = g & 31;
  if (e >= E) return;
  int s = src[e], d = dst[e];
  float coef = dinv[s] * w[e] * dinv[d];
  for (int k = lane; k < F; k += 32)
    atomicAdd(&o[(size_t)d * F + k], coef * t[(size_t)s * F + k]);
}

// o_bf16 = relu(h + b)  (feeds next GEMM directly in bf16)
__global__ void bias_relu_bf16_kernel(const float* __restrict__ h,
                                      const float* __restrict__ b,
                                      unsigned short* __restrict__ o,
                                      int n, int F) {
  int idx = blockIdx.x * blockDim.x + threadIdx.x;
  if (idx < n * F) {
    float v = h[idx] + b[idx % F];
    o[idx] = f2bf(v > 0.f ? v : 0.f);
  }
}

__global__ void bias_add_kernel(float* h, const float* __restrict__ b,
                                int n, int F) {
  int idx = blockIdx.x * blockDim.x + threadIdx.x;
  if (idx < n * F) h[idx] += b[idx % F];
}

// ---------------------------------------------------------------------------
// Launch
// ---------------------------------------------------------------------------
extern "C" void kernel_launch(void* const* d_in, const int* in_sizes, int n_in,
                              void* d_out, int out_size, void* d_ws,
                              size_t ws_size, hipStream_t stream) {
  const float* x    = (const float*)d_in[0];
  const float* xout = (const float*)d_in[1];
  const float* c    = (const float*)d_in[2];
  const int*   ei   = (const int*)d_in[3];   // [2,E]: src row then dst row
  const float* ew   = (const float*)d_in[4];
  const float* Wa   = (const float*)d_in[5];
  const float* Wc   = (const float*)d_in[6];
  const float* Wl   = (const float*)d_in[7];
  const float* bl   = (const float*)d_in[8];
  const float* W1   = (const float*)d_in[9];
  const float* b1   = (const float*)d_in[10];
  const float* W3   = (const float*)d_in[11];
  const float* b3   = (const float*)d_in[12];
  float* out = (float*)d_out;

  const int N   = in_sizes[0];
  const int E   = in_sizes[4];
  const int LM  = in_sizes[8];    // 1024
  const int H   = in_sizes[10];   // 128
  const int OUT = in_sizes[12];   // 64

  // ---- workspace carve-out (region0 overlaid after GEMM1) ----
  char* ws = (char*)d_ws;
  unsigned short* xb = (unsigned short*)ws;              // [N,LM] bf16
  const size_t region0 = (size_t)N * LM * sizeof(unsigned short);
  unsigned short* h0 = (unsigned short*)(ws + region0);  // [N,LM] bf16
  size_t off = 2 * region0;
  unsigned short* Wlb = (unsigned short*)(ws + off); off += (size_t)LM * LM * 2;
  unsigned short* W1b = (unsigned short*)(ws + off); off += (size_t)LM * H  * 2;
  unsigned short* W3b = (unsigned short*)(ws + off); off += (size_t)H  * OUT* 2;
  float* dinv = (float*)(ws + off);
  // overlays inside region0 (x_out_bf16 is dead after GEMM1)
  size_t r0 = 0;
  float* t1 = (float*)(ws + r0);                 r0 += (size_t)N * H * 4;
  float* h1 = (float*)(ws + r0);                 r0 += (size_t)N * H * 4;
  unsigned short* h1b = (unsigned short*)(ws + r0); r0 += (size_t)N * H * 2;
  float* t2 = (float*)(ws + r0);

  const int mblk = (N + BM - 1) / BM;
  dim3 blk(256);

  // ---- one-time bf16 conversions (bandwidth passes) ----
  f32_to_bf16_kernel<<<(int)(((long long)N * LM / 4 + 255) / 256), 256, 0,
                       stream>>>(xout, xb, (long long)N * LM / 4);
  f32_to_bf16_kernel<<<(int)(((long long)LM * LM / 4 + 255) / 256), 256, 0,
                       stream>>>(Wl, Wlb, (long long)LM * LM / 4);
  f32_to_bf16_kernel<<<(int)(((long long)LM * H / 4 + 255) / 256), 256, 0,
                       stream>>>(W1, W1b, (long long)LM * H / 4);
  f32_to_bf16_kernel<<<(int)(((long long)H * OUT / 4 + 255) / 256), 256, 0,
                       stream>>>(W3, W3b, (long long)H * OUT / 4);

  // ---- degree^{-1/2} with self loops (deg starts at 1.0) ----
  set_const_kernel<<<(N + 255) / 256, 256, 0, stream>>>(dinv, 1.0f, N);
  deg_edges_kernel<<<(E + 255) / 256, 256, 0, stream>>>(ei + E, ew, dinv, E);
  deg_fin_kernel<<<(N + 255) / 256, 256, 0, stream>>>(dinv, N);

  // GEMM1: h0 = relu(x_out@Wl + x*Wa + c*Wc + bl) -> bf16
  wmma_gemm_bf16<true><<<dim3(LM / BN, mblk), blk, 0, stream>>>(
      xb, Wlb, N, LM, LM, x, c, Wa, Wc, bl, h0);

  // GEMM2: t1 = h0 @ W1 -> fp32
  wmma_gemm_bf16<false><<<dim3(H / BN, mblk), blk, 0, stream>>>(
      h0, W1b, N, LM, H, nullptr, nullptr, nullptr, nullptr, nullptr, t1);

  // aggregation 1: h1b = bf16(relu(segsum(norm * t1[src]) + b1))
  self_init_kernel<<<(N * H + 255) / 256, 256, 0, stream>>>(t1, dinv, h1, N, H);
  edge_agg_kernel<<<(E * 32 + 255) / 256, 256, 0, stream>>>(ei, ei + E, ew,
                                                            dinv, t1, h1, E, H);
  bias_relu_bf16_kernel<<<(N * H + 255) / 256, 256, 0, stream>>>(h1, b1, h1b,
                                                                 N, H);

  // GEMM3: t2 = h1 @ W3 -> fp32
  wmma_gemm_bf16<false><<<dim3(OUT / BN, mblk), blk, 0, stream>>>(
      h1b, W3b, N, H, OUT, nullptr, nullptr, nullptr, nullptr, nullptr, t2);

  // aggregation 2: out = segsum(norm * t2[src]) + b3
  self_init_kernel<<<(N * OUT + 255) / 256, 256, 0, stream>>>(t2, dinv, out, N,
                                                              OUT);
  edge_agg_kernel<<<(E * 32 + 255) / 256, 256, 0, stream>>>(ei, ei + E, ew,
                                                            dinv, t2, out, E,
                                                            OUT);
  bias_add_kernel<<<(N * OUT + 255) / 256, 256, 0, stream>>>(out, b3, N, OUT);
}
